// Attention_68685116998007
// MI455X (gfx1250) — matrix operand order repaired and checked by reading, not compiled, not run on hardware
//
#include <hip/hip_runtime.h>

// ---------------------------------------------------------------------------
// XCA (cross-covariance attention) fused kernel for gfx1250 (MI455X).
// One workgroup (256 threads = 8 wave32) per batch element; all GEMMs via
// v_wmma_f32_16x16x32_f16 with 4x3 register blocking on the 384x384 GEMMs.
// ---------------------------------------------------------------------------

#define DIM    384
#define HEADS  8
#define HD     48
#define NTOK   49      // 7*7 spatial tokens
#define NPAD   64      // tokens padded for WMMA K/N tiling
#define BATCH  2048
#define XCA_SCALE 0.14433756729740643f   // 48^-0.5

typedef __attribute__((ext_vector_type(16))) _Float16 v16h;
typedef __attribute__((ext_vector_type(8)))  float    v8f;

// ---- LDS layout (byte offsets) --------------------------------------------
#define XSTRIDE  392                       // f16 row stride for [64][DIM] pads
#define OFF_X    0                         // [64][392] f16: x, later o+vlocal (50176 B)
#define OFF_QT   50176                     // [384][64]  f16: q transposed      (49152 B)
#define OFF_KT   99328                     // [384][64]  f16: k transposed; reused as attn-tmp [8][48][64]
#define OFF_V2   148480                    // [8][64][64] f16: v as [head][tok][chan] (65536 B)
#define OFF_ATTN 214016                    // [8*48][48] f32: attention logits/probs  (73728 B)
#define OFF_W1   287744                    // [8][8] f32 talking-head 1
#define OFF_W2   288000                    // [8][8] f32 talking-head 2
#define OFF_B1   288256                    // [8] f32
#define OFF_B2   288288                    // [8] f32
#define SMEM_BYTES 288320

// ---- WMMA helpers (CDNA5 layouts, wave32) ---------------------------------
__device__ __forceinline__ v8f wmma16(v16h a, v16h b, v8f c) {
  // D = A(16x32 f16) * B(32x16 f16) + C(16x16 f32)
  return __builtin_amdgcn_wmma_f32_16x16x32_f16(false, a, false, b, (short)0, c,
                                                false, false);
}

// A fragment: rowk points at (row m = mbase + (lane&15)) * stride + kbase.
// Lane holds K = hb..hb+7 and hb+16..hb+23 where hb = (lane>=16 ? 8 : 0).
__device__ __forceinline__ v16h frag_a(const _Float16* rowk, int lane) {
  const _Float16* p = rowk + ((lane & 16) ? 8 : 0);
  v16h r;
  ((uint4*)&r)[0] = *(const uint4*)(p);
  ((uint4*)&r)[1] = *(const uint4*)(p + 16);
  return r;
}

// B fragment: rowk points at (col n = nbase + (lane&15)) * stride + kbase,
// where memory is K-contiguous along the row. Lane holds K = hb..hb+7 and
// hb+16..hb+23, hb = (lane>=16 ? 8 : 0).
__device__ __forceinline__ v16h frag_b(const _Float16* rowk, int lane) {
  const _Float16* p = rowk + ((lane & 16) ? 8 : 0);
  v16h r;
  ((uint4*)&r)[0] = *(const uint4*)(p);
  ((uint4*)&r)[1] = *(const uint4*)(p + 16);
  return r;
}

// ---- weight f32 -> f16 conversion ------------------------------------------
__global__ void xca_prep_w(const float* __restrict__ Wq, const float* __restrict__ Wk,
                           const float* __restrict__ Wv, const float* __restrict__ Wp,
                           _Float16* __restrict__ out) {
  int i = blockIdx.x * blockDim.x + threadIdx.x;
  if (i >= DIM * DIM) return;
  out[i]                 = (_Float16)Wq[i];
  out[DIM * DIM + i]     = (_Float16)Wk[i];
  out[2 * DIM * DIM + i] = (_Float16)Wv[i];
  out[3 * DIM * DIM + i] = (_Float16)Wp[i];
}

// ---- fused main kernel -----------------------------------------------------
__global__ void __launch_bounds__(256, 1)
xca_main(const float* __restrict__ x, const _Float16* __restrict__ gW,
         const float* __restrict__ bq, const float* __restrict__ bk,
         const float* __restrict__ bv, const float* __restrict__ Wvl,
         const float* __restrict__ bvl, const float* __restrict__ Wth1,
         const float* __restrict__ bth1, const float* __restrict__ Wth2,
         const float* __restrict__ bth2, const float* __restrict__ bp,
         float* __restrict__ out) {
  extern __shared__ __align__(16) char smem[];
  _Float16* sX    = (_Float16*)(smem + OFF_X);     // [64][392]
  _Float16* sQT   = (_Float16*)(smem + OFF_QT);    // [384][64]
  _Float16* sKT   = (_Float16*)(smem + OFF_KT);    // [384][64]
  _Float16* sTmp  = (_Float16*)(smem + OFF_KT);    // [8][48][64] (aliases sKT)
  _Float16* sV2   = (_Float16*)(smem + OFF_V2);    // [8][64][64]
  _Float16* sO    = (_Float16*)(smem + OFF_X);     // [64][392]  (aliases sX)
  float*    sAttn = (float*)(smem + OFF_ATTN);     // [384][48]
  float*    sW1   = (float*)(smem + OFF_W1);
  float*    sW2   = (float*)(smem + OFF_W2);
  float*    sB1   = (float*)(smem + OFF_B1);
  float*    sB2   = (float*)(smem + OFF_B2);

  const int tid  = threadIdx.x;
  const int lane = tid & 31;
  const int wv_  = tid >> 5;         // wave id 0..7
  const int b    = blockIdx.x;
  const v8f vzero = {};

  // ---- phase 0: zero pads, load x -> f16, load talking-head consts --------
  for (int i = tid; i < (NPAD - NTOK) * XSTRIDE; i += 256)
    sX[NTOK * XSTRIDE + i] = (_Float16)0.0f;                 // token pad rows
  for (int i = tid; i < HEADS * NPAD * NPAD; i += 256)
    sV2[i] = (_Float16)0.0f;                                 // v buffer (pads)

  const float* xb = x + (size_t)b * NTOK * DIM;
  for (int i = tid; i < NTOK * (DIM / 4); i += 256) {        // float4 loads
    float4 f = ((const float4*)xb)[i];
    int e = i * 4, t = e / DIM, c = e % DIM;
    _Float16* p = sX + t * XSTRIDE + c;
    p[0] = (_Float16)f.x; p[1] = (_Float16)f.y;
    p[2] = (_Float16)f.z; p[3] = (_Float16)f.w;
  }
  if (tid < 64) { sW1[tid] = Wth1[tid]; sW2[tid] = Wth2[tid]; }
  if (tid < 8)  { sB1[tid] = bth1[tid]; sB2[tid] = bth2[tid]; }
  __syncthreads();

  // ---- phase 1: GEMM1  q/k/v = x @ W^T + b  (M=64 tok, N=384, K=384) ------
  // Register-blocked 4x3: each job computes a 64x48 C block.
  // 3 weights * 8 column groups = 24 jobs over 8 waves.
  for (int job = wv_; job < 24; job += 8) {
    int widx = job / 8, ng = job % 8;
    const _Float16* W    = gW + widx * DIM * DIM;
    const float*    bias = (widx == 0) ? bq : (widx == 1) ? bk : bv;
    const _Float16* arow0 = sX + (lane & 15) * XSTRIDE;
    const _Float16* brow0 = W + (ng * 48 + (lane & 15)) * DIM;

    v8f acc[4][3];
#pragma unroll
    for (int mt = 0; mt < 4; ++mt)
#pragma unroll
      for (int j = 0; j < 3; ++j) acc[mt][j] = vzero;

#pragma unroll 2
    for (int kt = 0; kt < 12; ++kt) {
      v16h a[4], bf[3];
#pragma unroll
      for (int mt = 0; mt < 4; ++mt)
        a[mt] = frag_a(arow0 + mt * 16 * XSTRIDE + kt * 32, lane);
#pragma unroll
      for (int j = 0; j < 3; ++j)
        bf[j] = frag_b(brow0 + j * 16 * DIM + kt * 32, lane);
#pragma unroll
      for (int mt = 0; mt < 4; ++mt)
#pragma unroll
        for (int j = 0; j < 3; ++j)
          acc[mt][j] = wmma16(a[mt], bf[j], acc[mt][j]);
    }

#pragma unroll
    for (int j = 0; j < 3; ++j) {
      int o    = (ng * 3 + j) * 16 + (lane & 15);
      float bo = bias[o];
      if (widx < 2) {
        _Float16* dst = (widx == 0 ? sQT : sKT) + o * NPAD;    // [chan][token]
#pragma unroll
        for (int mt = 0; mt < 4; ++mt) {
          int tbase = mt * 16 + ((lane & 16) ? 8 : 0);
#pragma unroll
          for (int r = 0; r < 8; ++r) {
            int t = tbase + r;
            dst[t] = (_Float16)((t < NTOK) ? (acc[mt][j][r] + bo) : 0.0f);
          }
        }
      } else {
        int h = o / HD, d = o % HD;
        _Float16* dst = sV2 + (h * NPAD) * NPAD + d;           // [head][tok][chan]
#pragma unroll
        for (int mt = 0; mt < 4; ++mt) {
          int tbase = mt * 16 + ((lane & 16) ? 8 : 0);
#pragma unroll
          for (int r = 0; r < 8; ++r) {
            int t = tbase + r;
            dst[t * NPAD] = (_Float16)((t < NTOK) ? (acc[mt][j][r] + bo) : 0.0f);
          }
        }
      }
    }
  }
  __syncthreads();

  // ---- phase 2: L2-normalize q,k rows (fold SCALE into q); depthwise 3x3 --
  for (int rj = tid; rj < 2 * DIM; rj += 256) {
    _Float16* row = (rj < DIM) ? (sQT + rj * NPAD) : (sKT + (rj - DIM) * NPAD);
    float s = 0.0f;
    for (int t = 0; t < NTOK; ++t) { float v = (float)row[t]; s += v * v; }
    float nrm = fmaxf(sqrtf(s), 1e-12f);
    float sc  = ((rj < DIM) ? XCA_SCALE : 1.0f) / nrm;
    for (int t = 0; t < NTOK; ++t) row[t] = (_Float16)((float)row[t] * sc);
  }
  // v_local (depthwise 3x3 'SAME' on v map) into sO (x is dead now)
  for (int i = tid; i < NTOK * DIM; i += 256) {
    int t = i / DIM, o = i % DIM;
    int y = t / 7, xx = t % 7, h = o / HD, d = o % HD;
    float acc = bvl[o];
#pragma unroll
    for (int dy = -1; dy <= 1; ++dy)
#pragma unroll
      for (int dx = -1; dx <= 1; ++dx) {
        int yy = y + dy, xc = xx + dx;
        if (yy >= 0 && yy < 7 && xc >= 0 && xc < 7) {
          float w = Wvl[((dy + 1) * 3 + (dx + 1)) * DIM + o];
          acc += w * (float)sV2[(h * NPAD + (yy * 7 + xc)) * NPAD + d];
        }
      }
    sO[t * XSTRIDE + o] = (_Float16)acc;
  }
  __syncthreads();

  // ---- phase 3: attn = q' @ k'^T per head  (M=N=48, K=64pad) --------------
  {
    int h = wv_;  // one head per wave
#pragma unroll
    for (int mt = 0; mt < 3; ++mt)
#pragma unroll
      for (int nt = 0; nt < 3; ++nt) {
        const _Float16* arow = sQT + (h * HD + mt * 16 + (lane & 15)) * NPAD;
        const _Float16* brow = sKT + (h * HD + nt * 16 + (lane & 15)) * NPAD;
        v8f acc = vzero;
#pragma unroll
        for (int kt = 0; kt < 2; ++kt)
          acc = wmma16(frag_a(arow + kt * 32, lane), frag_b(brow + kt * 32, lane), acc);
        int d = nt * 16 + (lane & 15);
        int cb = mt * 16 + ((lane & 16) ? 8 : 0);
        float* dst = sAttn + (h * HD) * HD + d;
#pragma unroll
        for (int r = 0; r < 8; ++r) dst[(cb + r) * HD] = acc[r];
      }
  }
  __syncthreads();

  // ---- phase 4a: talking-heads mix 1 -> sTmp (f16, aliases dead sKT) ------
  for (int i = tid; i < HD * HD; i += 256) {
    int c = i / HD, d = i % HD;
    float av[8];
#pragma unroll
    for (int h = 0; h < 8; ++h) av[h] = sAttn[(h * HD + c) * HD + d];
#pragma unroll
    for (int g = 0; g < 8; ++g) {
      float m = sB1[g];
#pragma unroll
      for (int h = 0; h < 8; ++h) m += av[h] * sW1[g * 8 + h];
      sTmp[(g * HD + c) * NPAD + d] = (_Float16)m;
    }
  }
  __syncthreads();

  // ---- phase 4b: softmax over d (48) per (g,c) row -> sAttn (f32) ---------
  for (int row = tid; row < HEADS * HD; row += 256) {
    const _Float16* rw = sTmp + row * NPAD;
    float mx = -1e30f;
    for (int d = 0; d < HD; ++d) mx = fmaxf(mx, (float)rw[d]);
    float sum = 0.0f;
    for (int d = 0; d < HD; ++d) sum += __expf((float)rw[d] - mx);
    float inv = 1.0f / sum;
    float* dst = sAttn + row * HD;
    for (int d = 0; d < HD; ++d) dst[d] = __expf((float)rw[d] - mx) * inv;
  }
  __syncthreads();

  // ---- phase 4c: talking-heads mix 2 -> sTmp (f16, K-padded to 64) --------
  for (int i = tid; i < HD * HD; i += 256) {
    int c = i / HD, d = i % HD;
    float av[8];
#pragma unroll
    for (int h = 0; h < 8; ++h) av[h] = sAttn[(h * HD + c) * HD + d];
#pragma unroll
    for (int g = 0; g < 8; ++g) {
      float m = sB2[g];
#pragma unroll
      for (int h = 0; h < 8; ++h) m += av[h] * sW2[g * 8 + h];
      sTmp[(g * HD + c) * NPAD + d] = (_Float16)m;
    }
  }
  for (int i = tid; i < HEADS * HD * (NPAD - HD); i += 256) {  // zero K pad
    int gc = i / (NPAD - HD), d = HD + i % (NPAD - HD);
    sTmp[gc * NPAD + d] = (_Float16)0.0f;
  }
  __syncthreads();

  // ---- phase 5: o = attn2 @ v, += v_local, ReLU  (M=48, N=64tok, K=64) ----
  {
    int h = wv_;
#pragma unroll
    for (int mt = 0; mt < 3; ++mt)
#pragma unroll
      for (int nt = 0; nt < 4; ++nt) {
        const _Float16* arow = sTmp + (h * HD + mt * 16 + (lane & 15)) * NPAD;
        const _Float16* brow = sV2 + (h * NPAD + nt * 16 + (lane & 15)) * NPAD;
        v8f acc = vzero;
#pragma unroll
        for (int kt = 0; kt < 2; ++kt)
          acc = wmma16(frag_a(arow + kt * 32, lane), frag_b(brow + kt * 32, lane), acc);
        int t  = nt * 16 + (lane & 15);
        int cb = mt * 16 + ((lane & 16) ? 8 : 0);
#pragma unroll
        for (int r = 0; r < 8; ++r) {
          int o = h * HD + cb + r;
          _Float16* p = sO + t * XSTRIDE + o;
          float v = acc[r] + (float)(*p);              // + v_local
          *p = (_Float16)fmaxf(v, 0.0f);               // ReLU, in place
        }
      }
  }
  __syncthreads();

  // ---- phase 6: out = relu(o) @ Wp^T + bp  (M=64 tok, N=384, K=384) -------
  // Register-blocked 4x3: one 64x48 C block per wave (8 column groups).
  {
    const _Float16* Wp16 = gW + 3 * DIM * DIM;
    float* outb = out + (size_t)b * NTOK * DIM;
    int ng = wv_;
    const _Float16* arow0 = sO + (lane & 15) * XSTRIDE;
    const _Float16* brow0 = Wp16 + (ng * 48 + (lane & 15)) * DIM;

    v8f acc[4][3];
#pragma unroll
    for (int mt = 0; mt < 4; ++mt)
#pragma unroll
      for (int j = 0; j < 3; ++j) acc[mt][j] = vzero;

#pragma unroll 2
    for (int kt = 0; kt < 12; ++kt) {
      v16h a[4], bf[3];
#pragma unroll
      for (int mt = 0; mt < 4; ++mt)
        a[mt] = frag_a(arow0 + mt * 16 * XSTRIDE + kt * 32, lane);
#pragma unroll
      for (int j = 0; j < 3; ++j)
        bf[j] = frag_b(brow0 + j * 16 * DIM + kt * 32, lane);
#pragma unroll
      for (int mt = 0; mt < 4; ++mt)
#pragma unroll
        for (int j = 0; j < 3; ++j)
          acc[mt][j] = wmma16(a[mt], bf[j], acc[mt][j]);
    }

#pragma unroll
    for (int j = 0; j < 3; ++j) {
      int o    = (ng * 3 + j) * 16 + (lane & 15);
      float bo = bp[o];
#pragma unroll
      for (int mt = 0; mt < 4; ++mt) {
        int tb = mt * 16 + ((lane & 16) ? 8 : 0);
#pragma unroll
        for (int r = 0; r < 8; ++r) {
          int t = tb + r;
          if (t < NTOK) outb[t * DIM + o] = acc[mt][j][r] + bo;
        }
      }
    }
  }
}

// ---------------------------------------------------------------------------
extern "C" void kernel_launch(void* const* d_in, const int* in_sizes, int n_in,
                              void* d_out, int out_size, void* d_ws, size_t ws_size,
                              hipStream_t stream) {
  const float* x    = (const float*)d_in[0];
  const float* Wq   = (const float*)d_in[1];
  const float* bq   = (const float*)d_in[2];
  const float* Wk   = (const float*)d_in[3];
  const float* bk   = (const float*)d_in[4];
  const float* Wv   = (const float*)d_in[5];
  const float* bv   = (const float*)d_in[6];
  const float* Wvl  = (const float*)d_in[7];
  const float* bvl  = (const float*)d_in[8];
  const float* Wth1 = (const float*)d_in[9];
  const float* bth1 = (const float*)d_in[10];
  const float* Wth2 = (const float*)d_in[11];
  const float* bth2 = (const float*)d_in[12];
  const float* Wp   = (const float*)d_in[13];
  const float* bp   = (const float*)d_in[14];
  _Float16* gW = (_Float16*)d_ws;   // 4 x 384x384 f16 = 1,179,648 B

  (void)in_sizes; (void)n_in; (void)out_size; (void)ws_size;

  hipFuncSetAttribute(reinterpret_cast<const void*>(xca_main),
                      hipFuncAttributeMaxDynamicSharedMemorySize, SMEM_BYTES);

  xca_prep_w<<<(DIM * DIM + 255) / 256, 256, 0, stream>>>(Wq, Wk, Wv, Wp, gW);
  xca_main<<<BATCH, 256, SMEM_BYTES, stream>>>(x, gW, bq, bk, bv, Wvl, bvl,
                                               Wth1, bth1, Wth2, bth2, bp,
                                               (float*)d_out);
}
